// BiLSTMCRF_13615046328794
// MI455X (gfx1250) — compile-verified, matrix operand order checked by reading, not run on hardware
//
#include <hip/hip_runtime.h>
#include <hip/hip_bf16.h>
#include <math.h>

// ---------------- types ----------------
typedef __attribute__((ext_vector_type(8)))  unsigned short v8u;
typedef __attribute__((ext_vector_type(16))) unsigned short v16u;
typedef __attribute__((ext_vector_type(16))) __bf16         v16bf;
typedef __attribute__((ext_vector_type(8)))  float          v8f;

#define Bv 32
#define Sv 256
#define Ev 256
#define Hv 256
#define Tv 32
#define MROWS (Sv * Bv)      // 8192
#define NGATE (4 * Hv)       // 1024
#define NBOTH (2 * NGATE)    // 2048 (both directions)

static __device__ __forceinline__ unsigned short f2bf(float f) {
  unsigned int u = __float_as_uint(f);
  unsigned int r = u + 0x7FFFu + ((u >> 16) & 1u);
  return (unsigned short)(r >> 16);
}
static __device__ __forceinline__ float sigm(float x) {
  return 1.0f / (1.0f + __expf(-x));
}
// One 16-bit WMMA A/B fragment (documented layout): lane l -> row/col l&15,
// elems j=0..7 at K=k0+8*(l>>4)+j, j=8..15 at K=k0+16+8*(l>>4)+(j-8)
// -> two contiguous 16-byte loads at p and p+16 (elements).
static __device__ __forceinline__ v16bf load_frag(const unsigned short* p) {
  v8u lo = *(const v8u*)(p);
  v8u hi = *(const v8u*)(p + 16);
  v16u t = __builtin_shufflevector(lo, hi, 0, 1, 2, 3, 4, 5, 6, 7,
                                   8, 9, 10, 11, 12, 13, 14, 15);
  return __builtin_bit_cast(v16bf, t);
}
static __device__ __forceinline__ v8f splat8(float v) {
  v8f z = {v, v, v, v, v, v, v, v};
  return z;
}

// ---------------- small prep kernels ----------------
__global__ void k_cvt_bf16(const float* __restrict__ s, unsigned short* __restrict__ d, int n) {
  for (int i = blockIdx.x * blockDim.x + threadIdx.x; i < n; i += gridDim.x * blockDim.x)
    d[i] = f2bf(s[i]);
}
__global__ void k_add_bias(const float* __restrict__ a, const float* __restrict__ b,
                           float* __restrict__ o, int n) {
  int i = blockIdx.x * blockDim.x + threadIdx.x;
  if (i < n) o[i] = a[i] + b[i];
}
__global__ void k_mask(const int* __restrict__ sent, float* __restrict__ mout) {
  int i = blockIdx.x * blockDim.x + threadIdx.x;
  if (i < Bv * Sv) mout[i] = (sent[i] != -1) ? 1.0f : 0.0f;
}

// ---------------- embedding + layernorm -> bf16 time-major [S][B][E] ----------------
__global__ void k_embed_ln(const int* __restrict__ sent, const float* __restrict__ table,
                           const float* __restrict__ gam, const float* __restrict__ bet,
                           unsigned short* __restrict__ xout) {
  __shared__ float red[Ev];
  int bid = blockIdx.x;            // b*S + s
  int b = bid >> 8, s = bid & 255;
  int e = threadIdx.x;
  int tok = sent[bid];
  float v = (tok >= 0) ? table[(long)tok * Ev + e] : 0.0f;
  red[e] = v; __syncthreads();
  for (int off = 128; off > 0; off >>= 1) { if (e < off) red[e] += red[e + off]; __syncthreads(); }
  float mean = red[0] * (1.0f / Ev);
  __syncthreads();
  float d = v - mean;
  red[e] = d * d; __syncthreads();
  for (int off = 128; off > 0; off >>= 1) { if (e < off) red[e] += red[e + off]; __syncthreads(); }
  float var = red[0] * (1.0f / Ev);
  float y = d * rsqrtf(var + 1e-5f) * gam[e] + bet[e];
  xout[((long)(s * Bv + b)) * Ev + e] = f2bf(y);
}

// ---------------- bf16 WMMA GEMM: C[M,N] = A[M,K] * W[N,K]^T + bias[N] ----------------
// K is a template constant so the register-double-buffered pipeline fully
// unrolls: the buffer rotation is absorbed by SSA renaming (no v_mov churn).
// block = 256 thr (8 waves); block tile = 32M x 512N; wave = 2 M x 4 N tiles.
template <int K>
__global__ __launch_bounds__(256) void k_gemm_bf16(const unsigned short* __restrict__ A,
                                                   const unsigned short* __restrict__ W,
                                                   const float* __restrict__ bias,
                                                   float* __restrict__ C, int N) {
  const int lane = threadIdx.x & 31, wave = threadIdx.x >> 5;
  const int nl = lane & 15, g = lane >> 4;
  const int mbase = blockIdx.x * 32;
  const int nbase = blockIdx.y * 512 + wave * 64;

  v8f acc[2][4];
  #pragma unroll
  for (int t = 0; t < 4; ++t) {
    float bb = bias[nbase + t * 16 + nl];
    acc[0][t] = splat8(bb);
    acc[1][t] = splat8(bb);
  }
  const unsigned short* ar0 = A + (long)(mbase + nl) * K + 8 * g;
  const unsigned short* ar1 = A + (long)(mbase + 16 + nl) * K + 8 * g;
  const unsigned short* wr0 = W + (long)(nbase + nl) * K + 8 * g;

  v16bf a_c0 = load_frag(ar0);
  v16bf a_c1 = load_frag(ar1);
  v16bf b_c[4];
  #pragma unroll
  for (int t = 0; t < 4; ++t) b_c[t] = load_frag(wr0 + (long)t * 16 * K);

  #pragma unroll
  for (int k0 = 0; k0 < K - 32; k0 += 32) {
    const int k1 = k0 + 32;
    v16bf a_n0 = load_frag(ar0 + k1);
    v16bf a_n1 = load_frag(ar1 + k1);
    v16bf b_n[4];
    #pragma unroll
    for (int t = 0; t < 4; ++t) b_n[t] = load_frag(wr0 + (long)t * 16 * K + k1);
    #pragma unroll
    for (int t = 0; t < 4; ++t) {
      acc[0][t] = __builtin_amdgcn_wmma_f32_16x16x32_bf16(false, a_c0, false, b_c[t],
                                                          (short)0, acc[0][t], false, false);
      acc[1][t] = __builtin_amdgcn_wmma_f32_16x16x32_bf16(false, a_c1, false, b_c[t],
                                                          (short)0, acc[1][t], false, false);
    }
    a_c0 = a_n0; a_c1 = a_n1;
    #pragma unroll
    for (int t = 0; t < 4; ++t) b_c[t] = b_n[t];
  }
  #pragma unroll
  for (int t = 0; t < 4; ++t) {   // peeled final K iteration
    acc[0][t] = __builtin_amdgcn_wmma_f32_16x16x32_bf16(false, a_c0, false, b_c[t],
                                                        (short)0, acc[0][t], false, false);
    acc[1][t] = __builtin_amdgcn_wmma_f32_16x16x32_bf16(false, a_c1, false, b_c[t],
                                                        (short)0, acc[1][t], false, false);
  }
  #pragma unroll
  for (int mt = 0; mt < 2; ++mt)
    #pragma unroll
    for (int t = 0; t < 4; ++t) {
      int ncol = nbase + t * 16 + nl;
      #pragma unroll
      for (int r = 0; r < 8; ++r) {
        int mrow = mbase + mt * 16 + r + 8 * g;   // C layout: lanes>=16 -> M=r+8
        C[(long)mrow * N + ncol] = acc[mt][t][r];
      }
    }
}

// ---------------- persistent LSTM recurrence (one block per direction) ----------------
// Gin: [S*B][2048] f32 (dir*1024 + gate*256 + h), Whh: bf16 [dir][1024][256],
// hout: bf16 [S*B][512] (dir*256 + h).
// Per step: hoist Gin reads (independent of h) before the WMMA K-loop so the
// HBM-streamed gate inputs overlap the recurrent GEMM.
// NOTE: the kk loop must stay a runtime loop (#pragma unroll 1) — full unroll
// lets LICM hoist the 32 step-invariant Whh fragments out of the step loop,
// which spills 1 KB/lane to scratch (observed in round 3).
__global__ __launch_bounds__(1024) void k_lstm(const float* __restrict__ Gin,
                                               const unsigned short* __restrict__ Whh,
                                               unsigned short* __restrict__ hout) {
  __shared__ unsigned short hl[Bv * Hv];   // 16 KB: h state, bf16
  __shared__ float          cl[Bv * Hv];   // 32 KB: c state, f32
  const int tid = threadIdx.x;
  const int dir = blockIdx.x;
  for (int i = tid; i < Bv * Hv; i += 1024) { hl[i] = 0; cl[i] = 0.0f; }
  __syncthreads();
  const int lane = tid & 31, wave = tid >> 5;
  const int nl = lane & 15, g = lane >> 4;
  const int m = wave & 1;          // M tile (batch 0-15 / 16-31)
  const int ht = wave >> 1;        // h tile (0..15)
  const int hcol = ht * 16 + nl;   // this lane's h column
  const unsigned short* Wd = Whh + (long)dir * NGATE * Hv + 8 * g;
  for (int it = 0; it < Sv; ++it) {
    const int s = (dir == 0) ? it : (Sv - 1 - it);
    // ---- issue Gin loads early; consumed after the GEMM ----
    float gin[4][8];
    #pragma unroll
    for (int r = 0; r < 8; ++r) {
      const int b = m * 16 + r + 8 * g;
      const long base = ((long)(s * Bv + b)) * NBOTH + (long)dir * NGATE + hcol;
      #pragma unroll
      for (int gt = 0; gt < 4; ++gt) gin[gt][r] = Gin[base + gt * Hv];
    }
    // ---- recurrent GEMM: gates += h_prev @ Whh^T ----
    v8f acc[4];
    #pragma unroll
    for (int t = 0; t < 4; ++t) acc[t] = splat8(0.0f);
    #pragma unroll 1
    for (int kk = 0; kk < 8; ++kk) {          // K = 256
      const int k0 = kk * 32;
      v16bf a = load_frag(hl + (m * 16 + nl) * Hv + k0 + 8 * g);
      v16bf bm[4];
      #pragma unroll
      for (int gt = 0; gt < 4; ++gt)
        bm[gt] = load_frag(Wd + (long)(gt * Hv + hcol) * Hv + k0);
      #pragma unroll
      for (int gt = 0; gt < 4; ++gt)
        acc[gt] = __builtin_amdgcn_wmma_f32_16x16x32_bf16(false, a, false, bm[gt],
                                                          (short)0, acc[gt], false, false);
    }
    __syncthreads();   // all waves done reading h before overwrite
    #pragma unroll
    for (int r = 0; r < 8; ++r) {
      const int b = m * 16 + r + 8 * g;
      float iv = acc[0][r] + gin[0][r];
      float fv = acc[1][r] + gin[1][r];
      float gv = acc[2][r] + gin[2][r];
      float ov = acc[3][r] + gin[3][r];
      float cn = sigm(fv) * cl[b * Hv + hcol] + sigm(iv) * tanhf(gv);
      float hn = sigm(ov) * tanhf(cn);
      cl[b * Hv + hcol] = cn;
      unsigned short hb = f2bf(hn);
      hl[b * Hv + hcol] = hb;
      hout[((long)(s * Bv + b)) * (2 * Hv) + dir * Hv + hcol] = hb;
    }
    __syncthreads();   // new h visible to next step
  }
}

// ---------------- output projection: emit = h1[8192,512] @ Wout[32,512]^T + b ----------------
// Each wave: 2 M-tiles x 2 N-tiles, fully unrolled double-buffered K loop.
// Writes time-major emit_ws [S][B][T] (for CRF) and batch-first d_out [B][S][T].
__global__ __launch_bounds__(256) void k_outproj(const unsigned short* __restrict__ A,
                                                 const unsigned short* __restrict__ W,
                                                 const float* __restrict__ bias,
                                                 float* __restrict__ emit_ws,
                                                 float* __restrict__ out_em) {
  const int lane = threadIdx.x & 31, wave = threadIdx.x >> 5;
  const int nl = lane & 15, g = lane >> 4;
  const int mbase = (blockIdx.x * 8 + wave) * 32;
  v8f acc[2][2];
  #pragma unroll
  for (int t = 0; t < 2; ++t) {
    float bb = bias[t * 16 + nl];
    acc[0][t] = splat8(bb);
    acc[1][t] = splat8(bb);
  }
  const unsigned short* ar0 = A + (long)(mbase + nl) * 512 + 8 * g;
  const unsigned short* ar1 = A + (long)(mbase + 16 + nl) * 512 + 8 * g;
  const unsigned short* wr0 = W + (long)nl * 512 + 8 * g;

  v16bf a_c0 = load_frag(ar0);
  v16bf a_c1 = load_frag(ar1);
  v16bf b_c[2];
  #pragma unroll
  for (int t = 0; t < 2; ++t) b_c[t] = load_frag(wr0 + (long)t * 16 * 512);

  #pragma unroll
  for (int k0 = 0; k0 < 512 - 32; k0 += 32) {
    const int k1 = k0 + 32;
    v16bf a_n0 = load_frag(ar0 + k1);
    v16bf a_n1 = load_frag(ar1 + k1);
    v16bf b_n[2];
    #pragma unroll
    for (int t = 0; t < 2; ++t) b_n[t] = load_frag(wr0 + (long)t * 16 * 512 + k1);
    #pragma unroll
    for (int t = 0; t < 2; ++t) {
      acc[0][t] = __builtin_amdgcn_wmma_f32_16x16x32_bf16(false, a_c0, false, b_c[t],
                                                          (short)0, acc[0][t], false, false);
      acc[1][t] = __builtin_amdgcn_wmma_f32_16x16x32_bf16(false, a_c1, false, b_c[t],
                                                          (short)0, acc[1][t], false, false);
    }
    a_c0 = a_n0; a_c1 = a_n1;
    b_c[0] = b_n[0]; b_c[1] = b_n[1];
  }
  #pragma unroll
  for (int t = 0; t < 2; ++t) {
    acc[0][t] = __builtin_amdgcn_wmma_f32_16x16x32_bf16(false, a_c0, false, b_c[t],
                                                        (short)0, acc[0][t], false, false);
    acc[1][t] = __builtin_amdgcn_wmma_f32_16x16x32_bf16(false, a_c1, false, b_c[t],
                                                        (short)0, acc[1][t], false, false);
  }
  #pragma unroll
  for (int mt = 0; mt < 2; ++mt)
    #pragma unroll
    for (int t = 0; t < 2; ++t) {
      int col = t * 16 + nl;
      #pragma unroll
      for (int r = 0; r < 8; ++r) {
        int row = mbase + mt * 16 + r + 8 * g;   // row = s*32 + b
        int s = row >> 5, b = row & 31;
        float v = acc[mt][t][r];
        emit_ws[(long)row * Tv + col] = v;
        out_em[(long)b * (Sv * Tv) + s * Tv + col] = v;
      }
    }
}

// ---------------- CRF mean log-likelihood (single block, 1024 threads) ----------------
__global__ __launch_bounds__(1024) void k_crf(const float* __restrict__ emit,   // [S][B][T]
                                              const int* __restrict__ tags,    // [B][S]
                                              const int* __restrict__ sent,    // [B][S]
                                              const float* __restrict__ start,
                                              const float* __restrict__ endw,
                                              const float* __restrict__ trans,
                                              float* __restrict__ out_loss) {
  __shared__ float tr[Tv * Tv];
  __shared__ float a0[Bv * Tv];
  __shared__ float a1[Bv * Tv];
  __shared__ float numr[Bv];
  __shared__ float denr[Bv];
  const int tid = threadIdx.x;
  const int b = tid >> 5, j = tid & 31;
  tr[tid] = trans[tid];
  a0[tid] = start[j] + emit[(long)(0 * Bv + b) * Tv + j];
  __syncthreads();
  // numerator: gold path score (one thread per batch element)
  if (tid < Bv) {
    const int bb = tid;
    int prev = tags[bb * Sv + 0];
    float num = start[prev] + emit[(long)(0 * Bv + bb) * Tv + prev];
    int last = prev;
    for (int t = 1; t < Sv; ++t) {
      int tg = tags[bb * Sv + t];
      float mval = (sent[bb * Sv + t] != -1) ? 1.0f : 0.0f;
      num += mval * (tr[prev * Tv + tg] + emit[(long)(t * Bv + bb) * Tv + tg]);
      if (mval > 0.0f) last = tg;
      prev = tg;
    }
    numr[bb] = num + endw[last];
  }
  // denominator: forward algorithm, double-buffered alpha
  float* cur = a0;
  float* nxt = a1;
  for (int t = 1; t < Sv; ++t) {
    float mval = (sent[b * Sv + t] != -1) ? 1.0f : 0.0f;
    float mx = -3.4e38f;
    #pragma unroll 8
    for (int i = 0; i < Tv; ++i) mx = fmaxf(mx, cur[b * Tv + i] + tr[i * Tv + j]);
    float ss = 0.0f;
    #pragma unroll 8
    for (int i = 0; i < Tv; ++i) ss += __expf(cur[b * Tv + i] + tr[i * Tv + j] - mx);
    float val = mx + __logf(ss) + emit[(long)(t * Bv + b) * Tv + j];
    nxt[b * Tv + j] = (mval > 0.0f) ? val : cur[b * Tv + j];
    __syncthreads();
    float* tmp = cur; cur = nxt; nxt = tmp;
  }
  if (tid < Bv) {
    const int bb = tid;
    float mx = -3.4e38f;
    for (int jj = 0; jj < Tv; ++jj) mx = fmaxf(mx, cur[bb * Tv + jj] + endw[jj]);
    float ss = 0.0f;
    for (int jj = 0; jj < Tv; ++jj) ss += __expf(cur[bb * Tv + jj] + endw[jj] - mx);
    denr[bb] = mx + __logf(ss);
  }
  __syncthreads();
  if (tid == 0) {
    float acc = 0.0f;
    for (int bb = 0; bb < Bv; ++bb) acc += numr[bb] - denr[bb];
    out_loss[0] = acc * (1.0f / Bv);
  }
}

// ---------------- host launcher ----------------
extern "C" void kernel_launch(void* const* d_in, const int* in_sizes, int n_in,
                              void* d_out, int out_size, void* d_ws, size_t ws_size,
                              hipStream_t stream) {
  const int*   sent   = (const int*)d_in[0];
  const int*   tags   = (const int*)d_in[2];
  const float* table  = (const float*)d_in[3];
  const float* lng    = (const float*)d_in[4];
  const float* lnb    = (const float*)d_in[5];
  const float* Wih0   = (const float*)d_in[6];
  const float* Whh0   = (const float*)d_in[7];
  const float* bih0   = (const float*)d_in[8];
  const float* bhh0   = (const float*)d_in[9];
  const float* Wih1   = (const float*)d_in[10];
  const float* Whh1   = (const float*)d_in[11];
  const float* bih1   = (const float*)d_in[12];
  const float* bhh1   = (const float*)d_in[13];
  const float* Wout   = (const float*)d_in[14];
  const float* bout   = (const float*)d_in[15];
  const float* cstart = (const float*)d_in[16];
  const float* cend   = (const float*)d_in[17];
  const float* ctrans = (const float*)d_in[18];

  float* out = (float*)d_out;
  float* out_emit = out;                       // [B][S][T] : 262144
  float* out_loss = out + Bv * Sv * Tv;        // 1
  float* out_mask = out_loss + 1;              // [B][S] : 8192

  char* ws = (char*)d_ws;
  const size_t MB = 1u << 20;
  unsigned short* x_bf    = (unsigned short*)(ws + 0);          //  4 MB [S*B][256]
  unsigned short* h0_bf   = (unsigned short*)(ws + 4 * MB);     //  8 MB [S*B][512]
  unsigned short* h1_bf   = (unsigned short*)(ws + 12 * MB);    //  8 MB [S*B][512]
  float*          Gin     = (float*)(ws + 20 * MB);             // 64 MB [S*B][2048]
  float*          emit_ws = (float*)(ws + 84 * MB);             //  1 MB [S*B][32]
  unsigned short* wih0_bf = (unsigned short*)(ws + 85 * MB);    //  1 MB [2048][256]
  unsigned short* whh0_bf = (unsigned short*)(ws + 86 * MB);    //  1 MB [2][1024][256]
  unsigned short* wih1_bf = (unsigned short*)(ws + 87 * MB);    //  2 MB [2048][512]
  unsigned short* whh1_bf = (unsigned short*)(ws + 89 * MB);    //  1 MB
  unsigned short* wout_bf = (unsigned short*)(ws + 90 * MB);    // 32 KB [32][512]
  float*          bias0   = (float*)(ws + 90 * MB + (64u << 10)); // 8 KB [2048]
  float*          bias1   = bias0 + NBOTH;                        // 8 KB

  // --- weight conversion to bf16 ---
  k_cvt_bf16<<<2048, 256, 0, stream>>>(Wih0, wih0_bf, NBOTH * Ev);
  k_cvt_bf16<<<2048, 256, 0, stream>>>(Whh0, whh0_bf, NBOTH * Hv);
  k_cvt_bf16<<<2048, 256, 0, stream>>>(Wih1, wih1_bf, NBOTH * 2 * Hv);
  k_cvt_bf16<<<2048, 256, 0, stream>>>(Whh1, whh1_bf, NBOTH * Hv);
  k_cvt_bf16<<<64, 256, 0, stream>>>(Wout, wout_bf, Tv * 2 * Hv);
  k_add_bias<<<8, 256, 0, stream>>>(bih0, bhh0, bias0, NBOTH);
  k_add_bias<<<8, 256, 0, stream>>>(bih1, bhh1, bias1, NBOTH);

  // --- embedding + layernorm (time-major bf16) ---
  k_embed_ln<<<Bv * Sv, Ev, 0, stream>>>(sent, table, lng, lnb, x_bf);

  // --- layer 0: input projection GEMM then recurrence ---
  k_gemm_bf16<Ev><<<dim3(MROWS / 32, NBOTH / 512), 256, 0, stream>>>(x_bf, wih0_bf, bias0,
                                                                     Gin, NBOTH);
  k_lstm<<<2, 1024, 0, stream>>>(Gin, whh0_bf, h0_bf);

  // --- layer 1 (input 2H=512), reusing Gin buffer ---
  k_gemm_bf16<2 * Hv><<<dim3(MROWS / 32, NBOTH / 512), 256, 0, stream>>>(h0_bf, wih1_bf, bias1,
                                                                         Gin, NBOTH);
  k_lstm<<<2, 1024, 0, stream>>>(Gin, whh1_bf, h1_bf);

  // --- output projection + outputs ---
  k_outproj<<<MROWS / 256, 256, 0, stream>>>(h1_bf, wout_bf, bout, emit_ws, out_emit);
  k_mask<<<Bv * Sv / 256, 256, 0, stream>>>(sent, out_mask);
  k_crf<<<1, 1024, 0, stream>>>(emit_ws, tags, sent, cstart, cend, ctrans, out_loss);
}
